// MultiHead_86990267613189
// MI455X (gfx1250) — compile-verified, matrix operand order checked by reading, not run on hardware
//
#include <hip/hip_runtime.h>
#include <cstdint>

typedef __attribute__((ext_vector_type(16))) _Float16     v16h;
typedef __attribute__((ext_vector_type(8)))  float        v8f;
typedef __attribute__((ext_vector_type(4)))  unsigned int v4u;
typedef __attribute__((ext_vector_type(8)))  int          v8i;
typedef __attribute__((ext_vector_type(4)))  int          v4i;

#define B_   4
#define S_   2048
#define D_   512
#define H_   8
#define DK_  64
#define M_   (B_ * S_)          // 8192 rows
#define BSTRIDE_ 528            // LDS row stride in halfs (512 + 8-dword TDM pad)

static __device__ inline v8f wmma_f16(v16h a, v16h b, v8f c) {
    // D = A(16x32 f16) * B(32x16 f16) + C(16x16 f32)
    return __builtin_amdgcn_wmma_f32_16x16x32_f16(
        /*neg_a=*/false, a, /*neg_b=*/false, b,
        /*c_mod=*/(short)0, c, /*reuse_a=*/false, /*reuse_b=*/false);
}

// ---------------------------------------------------------------------------
// Tensor Data Mover: DMA a 64-row x 512-half (N-major) weight slab into LDS.
// 2-D descriptor per CDNA5 ISA.
// LDS padding: 8 DWORDs every 256 DWORDs -> row stride 528 halfs (1056 B,
// 32B-aligned, 8-bank rotation per row to break ds bank conflicts).
// This toolchain's builtin takes 6 args (g0, g1, g2, g3, g4, cpol).
// ---------------------------------------------------------------------------
static __device__ inline void tdm_load_bslab(const _Float16* src,
                                             _Float16* dst_lds) {
    const unsigned long long ga = (unsigned long long)(uintptr_t)src;
    const unsigned int       la = (unsigned int)(uintptr_t)dst_lds; // LDS byte addr

    v4u g0;
    g0.x = 1u;                                    // count=1 user descriptor
    g0.y = la;                                    // lds_addr
    g0.z = (unsigned int)ga;                      // global_addr[31:0]
    g0.w = (unsigned int)(ga >> 32) | (2u << 30); // global_addr[56:32] | type=2

    v8i g1;
    g1[0] = (int)((1u << 16)      // data_size = 1 -> 2 bytes
                | (1u << 20)      // pad_enable
                | (7u << 22)      // pad_interval: 256 DWORDs
                | (7u << 25));    // pad_amount:   8 DWORDs
    g1[1] = (int)(D_ << 16);      // tensor_dim0 = 512     (bits 79:48 lo16)
    g1[2] = (int)(D_ << 16);      // dim0 hi | tensor_dim1 = 512 lo16
    g1[3] = (int)(D_ << 16);      // dim1 hi | tile_dim0 = 512
    g1[4] = 64;                   // tile_dim1 = 64 rows, tile_dim2 = 0
    g1[5] = (int)D_;              // tensor_dim0_stride = 512
    g1[6] = 0;                    // stride hi | tensor_dim1_stride lo = 0
    g1[7] = 0;

    const v4i zero4 = {0, 0, 0, 0};
    const v8i zero8 = {0, 0, 0, 0, 0, 0, 0, 0};
    __builtin_amdgcn_tensor_load_to_lds(g0, g1, zero4, zero4, zero8,
                                        /*cpol=*/0);
}

// ---------------------------------------------------------------------------
// 1) LayerNorm(q) fused with f32 -> f16 cast.  One block (256 thr) per row.
// ---------------------------------------------------------------------------
__global__ void ln_cast_kernel(const float* __restrict__ x,
                               const float* __restrict__ gamma,
                               const float* __restrict__ beta,
                               _Float16* __restrict__ out) {
    __shared__ float red[256];
    const int row = blockIdx.x;
    const int tid = threadIdx.x;
    const float* xr = x + (size_t)row * D_;
    const float v0 = xr[tid];
    const float v1 = xr[tid + 256];

    red[tid] = v0 + v1;
    __syncthreads();
    for (int w = 128; w > 0; w >>= 1) {
        if (tid < w) red[tid] += red[tid + w];
        __syncthreads();
    }
    const float mu = red[0] * (1.0f / D_);
    __syncthreads();

    const float d0 = v0 - mu, d1 = v1 - mu;
    red[tid] = d0 * d0 + d1 * d1;
    __syncthreads();
    for (int w = 128; w > 0; w >>= 1) {
        if (tid < w) red[tid] += red[tid + w];
        __syncthreads();
    }
    const float rstd = rsqrtf(red[0] * (1.0f / D_) + 1e-6f);

    _Float16* orow = out + (size_t)row * D_;
    orow[tid]       = (_Float16)(d0 * rstd * gamma[tid] + beta[tid]);
    orow[tid + 256] = (_Float16)(d1 * rstd * gamma[tid + 256] + beta[tid + 256]);
}

// ---------------------------------------------------------------------------
// 2) Elementwise f32 -> f16 cast (k, v).
// ---------------------------------------------------------------------------
__global__ void cast_f16_kernel(const float* __restrict__ in,
                                _Float16* __restrict__ out, int n) {
    const int i = blockIdx.x * blockDim.x + threadIdx.x;
    if (i < n) out[i] = (_Float16)in[i];
}

// ---------------------------------------------------------------------------
// 3) Weight cast + transpose (+ optional scale): WT[n][k] = W[k][n] * scale.
// ---------------------------------------------------------------------------
__global__ void castT_kernel(const float* __restrict__ W,
                             _Float16* __restrict__ WT, float scale) {
    const int i = blockIdx.x * blockDim.x + threadIdx.x;  // n*512 + k
    const int n = i >> 9;
    const int k = i & (D_ - 1);
    WT[i] = (_Float16)(W[(size_t)k * D_ + n] * scale);
}

// ---------------------------------------------------------------------------
// 4) Projection GEMM with TDM-staged weights (template removes the runtime
//    store-layout branch).  Block = 4 waves; TDM loads the block's 64-column
//    weight slab (64x512 f16, 66KB padded) into LDS once; block sweeps 8
//    row-tiles of 64 rows.  B fragments come from LDS (ds_load_b128), A
//    streams from global.  k-loop kept rolled so LICM cannot hoist B
//    fragments across row tiles (avoids scratch spills).
//    VTRANS=true stores V transposed as VhT[B][H][64][S].
// ---------------------------------------------------------------------------
template <bool VTRANS>
__global__ void gemm_proj_kernel(const _Float16* __restrict__ A,
                                 const _Float16* __restrict__ BT,
                                 _Float16* __restrict__ C) {
    __shared__ _Float16 bsl[64 * BSTRIDE_];
    const int lane = threadIdx.x & 31;
    const int wave = threadIdx.x >> 5;
    const int hf   = lane >> 4;
    const int nl   = lane & 15;
    const int colBase = blockIdx.y * 64;

    if (wave == 0) {
        tdm_load_bslab(BT + (size_t)colBase * D_, &bsl[0]);
        __builtin_amdgcn_s_wait_tensorcnt(0);
    }
    __syncthreads();

    for (int rt = 0; rt < 8; ++rt) {
        const int rowBase = (blockIdx.x * 8 + rt) * 64 + wave * 16;
        const _Float16* arow = A + (size_t)(rowBase + nl) * D_ + hf * 16;
        v8f acc[4] = {};

#pragma clang loop unroll(disable)
        for (int kk = 0; kk < D_; kk += 32) {
            v16h av = *(const v16h*)(arow + kk);
            __builtin_prefetch(arow + kk + 64, 0, 3);   // global_prefetch_b8
#pragma unroll
            for (int j = 0; j < 4; ++j) {
                const _Float16* bl =
                    bsl + (size_t)(j * 16 + nl) * BSTRIDE_ + kk + hf * 16;
                v16h bv = *(const v16h*)bl;
                acc[j] = wmma_f16(av, bv, acc[j]);
            }
        }

#pragma unroll
        for (int j = 0; j < 4; ++j) {
#pragma unroll
            for (int r = 0; r < 8; ++r) {
                const int m = rowBase + r + 8 * hf;
                const int n = colBase + j * 16 + nl;
                const _Float16 val = (_Float16)acc[j][r];
                if (!VTRANS) {
                    C[(size_t)m * D_ + n] = val;
                } else {
                    const int b = m >> 11;          // m / S_
                    const int s = m & (S_ - 1);
                    const int h = n >> 6;           // n / DK_
                    const int d = n & (DK_ - 1);
                    C[(((size_t)b * H_ + h) * DK_ + d) * S_ + s] = val;
                }
            }
        }
    }
}

// ---------------------------------------------------------------------------
// 5) Flash-attention: one wave per 16-row q tile, Sk in steps of 32.
//    Scores via WMMA, mask, online softmax (cross-lane shuffles), P staged
//    in LDS (C-layout -> A-layout), then O += P * V via WMMA vs transposed V.
// ---------------------------------------------------------------------------
__global__ void attn_kernel(const _Float16* __restrict__ Qh,
                            const _Float16* __restrict__ Kh,
                            const _Float16* __restrict__ VhT,
                            const int* __restrict__ mask,
                            _Float16* __restrict__ Ctx) {
    __shared__ _Float16 lds[4][16 * 32];   // one 16x32 P tile per wave

    const int lane = threadIdx.x & 31;
    const int wave = threadIdx.x >> 5;
    const int hf   = lane >> 4;
    const int nl   = lane & 15;

    const int tile = blockIdx.x * 4 + wave;     // 0 .. B*H*(S/16)-1
    const int qt   = tile & (S_ / 16 - 1);
    const int bh   = tile >> 7;
    const int h    = bh & (H_ - 1);
    const int b    = bh >> 3;
    const int qbase = qt * 16;

    const _Float16* qrow =
        Qh + (size_t)(b * S_ + qbase + nl) * D_ + h * DK_;
    const v16h qa0 = *(const v16h*)(qrow + hf * 16);        // dims  0..31
    const v16h qa1 = *(const v16h*)(qrow + 32 + hf * 16);   // dims 32..63

    const _Float16* kbase = Kh + (size_t)(b * S_) * D_ + h * DK_;
    const _Float16* vbase = VhT + ((size_t)b * H_ + h) * DK_ * S_;
    const int* mrow = mask + ((size_t)b * S_ + qbase) * S_;

    float mrun[8], lrun[8];
#pragma unroll
    for (int r = 0; r < 8; ++r) { mrun[r] = -1e30f; lrun[r] = 0.0f; }
    v8f o[4] = {};

    _Float16* pl = &lds[wave][0];

    for (int sk = 0; sk < S_; sk += 32) {
        v8f c0 = {}, c1 = {};
        {
            const _Float16* k0 = kbase + (size_t)(sk + nl) * D_ + hf * 16;
            v16h b0  = *(const v16h*)(k0);
            v16h b0b = *(const v16h*)(k0 + 32);
            c0 = wmma_f16(qa0, b0, c0);
            c0 = wmma_f16(qa1, b0b, c0);
            const _Float16* k1 = kbase + (size_t)(sk + 16 + nl) * D_ + hf * 16;
            v16h b1  = *(const v16h*)(k1);
            v16h b1b = *(const v16h*)(k1 + 32);
            c1 = wmma_f16(qa0, b1, c1);
            c1 = wmma_f16(qa1, b1b, c1);
        }

#pragma unroll
        for (int r = 0; r < 8; ++r) {
            const int qm = r + 8 * hf;
            const int* mr = mrow + (size_t)qm * S_;
            float x0 = c0[r];
            float x1 = c1[r];
            if (mr[sk + nl] == 0)      x0 = -1e9f;
            if (mr[sk + 16 + nl] == 0) x1 = -1e9f;

            float t = fmaxf(x0, x1);
            for (int off = 1; off < 16; off <<= 1)
                t = fmaxf(t, __shfl_xor(t, off, 32));
            const float newm = fmaxf(mrun[r], t);
            const float corr = __expf(mrun[r] - newm);
            const float p0 = __expf(x0 - newm);
            const float p1 = __expf(x1 - newm);
            float rs = p0 + p1;
            for (int off = 1; off < 16; off <<= 1)
                rs += __shfl_xor(rs, off, 32);
            lrun[r] = lrun[r] * corr + rs;
            mrun[r] = newm;

            o[0][r] *= corr;
            o[1][r] *= corr;
            o[2][r] *= corr;
            o[3][r] *= corr;

            pl[qm * 32 + nl]      = (_Float16)p0;
            pl[qm * 32 + 16 + nl] = (_Float16)p1;
        }

        const v16h pa = *(const v16h*)(pl + nl * 32 + hf * 16);
#pragma unroll
        for (int j = 0; j < 4; ++j) {
            const _Float16* vrow =
                vbase + (size_t)(j * 16 + nl) * S_ + sk + hf * 16;
            v16h vb = *(const v16h*)vrow;
            o[j] = wmma_f16(pa, vb, o[j]);
        }
    }

#pragma unroll
    for (int j = 0; j < 4; ++j) {
#pragma unroll
        for (int r = 0; r < 8; ++r) {
            const int qm = r + 8 * hf;
            const float val = o[j][r] / lrun[r];
            Ctx[(size_t)(b * S_ + qbase + qm) * D_ + h * DK_ + j * 16 + nl] =
                (_Float16)val;
        }
    }
}

// ---------------------------------------------------------------------------
// 6) Output GEMM + residual, same TDM-staged weight scheme, f32 out.
// ---------------------------------------------------------------------------
__global__ void gemm_fc_kernel(const _Float16* __restrict__ A,
                               const _Float16* __restrict__ BT,
                               const float* __restrict__ resid,
                               float* __restrict__ out) {
    __shared__ _Float16 bsl[64 * BSTRIDE_];
    const int lane = threadIdx.x & 31;
    const int wave = threadIdx.x >> 5;
    const int hf   = lane >> 4;
    const int nl   = lane & 15;
    const int colBase = blockIdx.y * 64;

    if (wave == 0) {
        tdm_load_bslab(BT + (size_t)colBase * D_, &bsl[0]);
        __builtin_amdgcn_s_wait_tensorcnt(0);
    }
    __syncthreads();

    for (int rt = 0; rt < 8; ++rt) {
        const int rowBase = (blockIdx.x * 8 + rt) * 64 + wave * 16;
        const _Float16* arow = A + (size_t)(rowBase + nl) * D_ + hf * 16;
        v8f acc[4] = {};

#pragma clang loop unroll(disable)
        for (int kk = 0; kk < D_; kk += 32) {
            v16h av = *(const v16h*)(arow + kk);
            __builtin_prefetch(arow + kk + 64, 0, 3);
#pragma unroll
            for (int j = 0; j < 4; ++j) {
                const _Float16* bl =
                    bsl + (size_t)(j * 16 + nl) * BSTRIDE_ + kk + hf * 16;
                v16h bv = *(const v16h*)bl;
                acc[j] = wmma_f16(av, bv, acc[j]);
            }
        }

#pragma unroll
        for (int j = 0; j < 4; ++j) {
#pragma unroll
            for (int r = 0; r < 8; ++r) {
                const int m = rowBase + r + 8 * hf;
                const int n = colBase + j * 16 + nl;
                out[(size_t)m * D_ + n] = acc[j][r] + resid[(size_t)m * D_ + n];
            }
        }
    }
}

// ---------------------------------------------------------------------------
extern "C" void kernel_launch(void* const* d_in, const int* in_sizes, int n_in,
                              void* d_out, int out_size, void* d_ws, size_t ws_size,
                              hipStream_t stream) {
    const float* q     = (const float*)d_in[0];
    const float* k     = (const float*)d_in[1];
    const float* v     = (const float*)d_in[2];
    const int*   mask  = (const int*)d_in[3];
    const float* Wq    = (const float*)d_in[4];
    const float* Wk    = (const float*)d_in[5];
    const float* Wv    = (const float*)d_in[6];
    const float* Wfc   = (const float*)d_in[7];
    const float* gamma = (const float*)d_in[8];
    const float* beta  = (const float*)d_in[9];
    float* out = (float*)d_out;

    char* ws = (char*)d_ws;
    size_t off = 0;
    auto take = [&](size_t bytes) -> char* {
        char* p = ws + off;
        off += (bytes + 255) & ~(size_t)255;
        return p;
    };
    _Float16* qn16  = (_Float16*)take((size_t)M_ * D_ * 2);
    _Float16* k16   = (_Float16*)take((size_t)M_ * D_ * 2);
    _Float16* v16   = (_Float16*)take((size_t)M_ * D_ * 2);
    _Float16* WqT   = (_Float16*)take((size_t)D_ * D_ * 2);
    _Float16* WkT   = (_Float16*)take((size_t)D_ * D_ * 2);
    _Float16* WvT   = (_Float16*)take((size_t)D_ * D_ * 2);
    _Float16* WfcT  = (_Float16*)take((size_t)D_ * D_ * 2);
    _Float16* Qh16  = (_Float16*)take((size_t)M_ * D_ * 2);
    _Float16* Kh16  = (_Float16*)take((size_t)M_ * D_ * 2);
    _Float16* VhT16 = (_Float16*)take((size_t)M_ * D_ * 2);
    _Float16* Ctx16 = (_Float16*)take((size_t)M_ * D_ * 2);

    ln_cast_kernel<<<M_, 256, 0, stream>>>(q, gamma, beta, qn16);

    const int nel = M_ * D_;
    cast_f16_kernel<<<nel / 256, 256, 0, stream>>>(k, k16, nel);
    cast_f16_kernel<<<nel / 256, 256, 0, stream>>>(v, v16, nel);

    const int nw = D_ * D_;
    castT_kernel<<<nw / 256, 256, 0, stream>>>(Wq, WqT, 0.125f);  // fold 1/sqrt(dk)
    castT_kernel<<<nw / 256, 256, 0, stream>>>(Wk, WkT, 1.0f);
    castT_kernel<<<nw / 256, 256, 0, stream>>>(Wv, WvT, 1.0f);
    castT_kernel<<<nw / 256, 256, 0, stream>>>(Wfc, WfcT, 1.0f);

    dim3 ggrid(M_ / 512, D_ / 64);   // 16 x 8 blocks, 8 row-tiles per block
    gemm_proj_kernel<false><<<ggrid, 128, 0, stream>>>(qn16, WqT, Qh16);
    gemm_proj_kernel<false><<<ggrid, 128, 0, stream>>>(k16,  WkT, Kh16);
    gemm_proj_kernel<true ><<<ggrid, 128, 0, stream>>>(v16,  WvT, VhT16);

    attn_kernel<<<(B_ * H_ * (S_ / 16)) / 4, 128, 0, stream>>>(
        Qh16, Kh16, VhT16, mask, Ctx16);

    gemm_fc_kernel<<<ggrid, 128, 0, stream>>>(Ctx16, WfcT, q, out);

    (void)in_sizes; (void)n_in; (void)out_size; (void)ws_size;
}
